// WindowNonLocalDenoising_18425409700243
// MI455X (gfx1250) — matrix operand order repaired and checked by reading, not compiled, not run on hardware
//
#include <hip/hip_runtime.h>
#include <hip/hip_bf16.h>

typedef __attribute__((ext_vector_type(16))) __bf16       v16bf;
typedef __attribute__((ext_vector_type(8)))  float        v8f;
typedef __attribute__((ext_vector_type(8)))  unsigned int v8u;

#define C_DIM   256
#define N_VALID 225
#define WSZ     15
#define HW      200
#define NT      15     // 240/16 n-tiles
#define ETL     9      // 144/16 e-tiles
#define CTL     16     // 256/16 c-tiles
#define KC      8      // K over C: 256/32
#define KE      5      // K over E: 160/32 (masked at 144)
#define KM      8      // K over M: 256/32 (masked at 240)

// bf16 weight offsets in d_ws (elements)
#define THW_OFF 0
#define PHW_OFF 36864          // 144*256
#define PRW_OFF 73728          // 2*144*256
#define W_TOTAL 139264         // + 256*256

// padded LDS row strides (elements): odd dword counts -> bank-conflict-free
#define XN_S   258     // Xn[240][258], Aff/OutT[240][258]  (129 dwords/row)
#define TT_S   146     // Tt[240][146], PhT[240][146]       (73 dwords/row)
#define X2_S   242     // Xs2[256][242]                     (121 dwords/row)

// LDS layout (bytes):
//  [0,      123840): Xn [240][258] (P0/P1)  /  Aff,OutT [240][258] (P2..P4)
//  [123840, 193920): Tt [240][146] (P1/P2)  \  Xs2 [256][242] -> ends 247744 (P3/P4)
//  [193920, 264000): PhT[240][146] (P1/P2)  /
//  +64B pad for masked-load overreads
#define SMEM_BYTES 264064

__device__ __forceinline__ v8f zero8() { v8f z = {0.f,0.f,0.f,0.f,0.f,0.f,0.f,0.f}; return z; }

__device__ __forceinline__ v8f wmma_bf16(v16bf a, v16bf b, v8f c) {
  return __builtin_amdgcn_wmma_f32_16x16x32_bf16(false, a, false, b, (short)0, c, false, false);
}

// A fragment (16x32 bf16): lane m = lane&15; dword j -> K = (lane>=16)*16 + 2j
__device__ __forceinline__ int afrag_k(int j, int hi) {
  return (hi << 4) + (j << 1);
}

// A natural from LDS: element (m,k) = base[(row0+m)*stride + col0+k]
__device__ __forceinline__ v16bf load_a_lds_nm(const __bf16* base, int stride, int row0, int col0) {
  const int lane = threadIdx.x & 31, m = lane & 15, hi = lane >> 4;
  const unsigned int* pu = (const unsigned int*)base;
  const int rowoff = (row0 + m) * stride;
  v8u u;
  #pragma unroll
  for (int j = 0; j < 8; ++j)
    u[j] = pu[(rowoff + col0 + afrag_k(j, hi)) >> 1];
  return __builtin_bit_cast(v16bf, u);
}

__device__ __forceinline__ v16bf load_a_lds_mk(const __bf16* base, int stride, int row0, int col0, int klim) {
  const int lane = threadIdx.x & 31, m = lane & 15, hi = lane >> 4;
  const unsigned int* pu = (const unsigned int*)base;
  const int rowoff = (row0 + m) * stride;
  v8u u;
  #pragma unroll
  for (int j = 0; j < 8; ++j) {
    int k = col0 + afrag_k(j, hi);
    unsigned int v = pu[(rowoff + k) >> 1];
    u[j] = (k < klim) ? v : 0u;
  }
  return __builtin_bit_cast(v16bf, u);
}

// A natural from global bf16: element (m,k) = g[(row0+m)*stride + col0+k] (2x b128 per lane)
__device__ __forceinline__ v16bf load_a_gbf(const __bf16* g, int stride, int row0, int col0) {
  const int lane = threadIdx.x & 31, m = lane & 15, hi = lane >> 4;
  const unsigned int* pu = (const unsigned int*)(g + (size_t)(row0 + m) * stride);
  v8u u;
  #pragma unroll
  for (int j = 0; j < 8; ++j)
    u[j] = pu[(col0 + afrag_k(j, hi)) >> 1];
  return __builtin_bit_cast(v16bf, u);
}

// B fragment (32x16 bf16): lane n = lane&15; K base kb = (lane>=16)*16; dword j -> K = kb+2j
// B from transposed LDS matrix: element (k,n) = base[(n0+n)*stride + k0+k]
__device__ __forceinline__ v16bf load_b_ldsT_nm(const __bf16* base, int stride, int k0, int n0) {
  const int lane = threadIdx.x & 31, n = lane & 15, kb = (lane >> 4) << 4;
  const unsigned int* pu = (const unsigned int*)base;
  const int rowoff = (n0 + n) * stride;
  v8u u;
  #pragma unroll
  for (int j = 0; j < 8; ++j)
    u[j] = pu[(rowoff + k0 + kb + 2*j) >> 1];
  return __builtin_bit_cast(v16bf, u);
}

__device__ __forceinline__ v16bf load_b_ldsT_mk(const __bf16* base, int stride, int k0, int n0, int klim) {
  const int lane = threadIdx.x & 31, n = lane & 15, kb = (lane >> 4) << 4;
  const unsigned int* pu = (const unsigned int*)base;
  const int rowoff = (n0 + n) * stride;
  v8u u;
  #pragma unroll
  for (int j = 0; j < 8; ++j) {
    int k = k0 + kb + 2*j;
    unsigned int v = pu[(rowoff + k) >> 1];
    u[j] = (k < klim) ? v : 0u;
  }
  return __builtin_bit_cast(v16bf, u);
}

// B from global bf16 stored [N][K]: element (k,n) = g[(n0+n)*stride + k0+k] (32B contiguous per lane)
__device__ __forceinline__ v16bf load_b_gbfT(const __bf16* g, int stride, int k0, int n0) {
  const int lane = threadIdx.x & 31, n = lane & 15, kb = (lane >> 4) << 4;
  const unsigned int* pu = (const unsigned int*)(g + (size_t)(n0 + n) * stride + k0 + kb);
  v8u u;
  #pragma unroll
  for (int j = 0; j < 8; ++j)
    u[j] = pu[j];
  return __builtin_bit_cast(v16bf, u);
}

// D tile (f32): element r -> row M = r + 8*(lane>=16), col N = lane&15
__device__ __forceinline__ void store_d_bf16(__bf16* base, int stride, int row0, int col0, v8f d) {
  const int lane = threadIdx.x & 31, l15 = lane & 15, hi = lane >> 4;
  #pragma unroll
  for (int r = 0; r < 8; ++r)
    base[(size_t)(row0 + r + 8*hi) * stride + col0 + l15] = (__bf16)d[r];
}

// transposed D store: element (M,N) -> base[N*stride + M]  (per-lane contiguous in M)
__device__ __forceinline__ void store_d_bf16_T(__bf16* base, int stride, int row0, int col0, v8f d) {
  const int lane = threadIdx.x & 31, l15 = lane & 15, hi = lane >> 4;
  __bf16* p = base + (size_t)(col0 + l15) * stride + row0 + 8*hi;
  #pragma unroll
  for (int r = 0; r < 8; ++r)
    p[r] = (__bf16)d[r];
}

// ---- prep: convert f32 weights -> bf16 in workspace (runs every call; deterministic) ----
__global__ void cvt_weights_kernel(const float* __restrict__ thw, const float* __restrict__ phw,
                                   const float* __restrict__ prw, __bf16* __restrict__ wb) {
  int i = blockIdx.x * blockDim.x + threadIdx.x;
  if (i < PHW_OFF)          wb[i] = (__bf16)thw[i];
  else if (i < PRW_OFF)     wb[i] = (__bf16)phw[i - PHW_OFF];
  else if (i < W_TOTAL)     wb[i] = (__bf16)prw[i - PRW_OFF];
}

__global__ __launch_bounds__(256, 1)
void wnl_attn_kernel(const float* __restrict__ x, const __bf16* __restrict__ wb,
                     float* __restrict__ out)
{
  __shared__ __align__(16) char smem[SMEM_BYTES];
  __bf16* Xn   = (__bf16*)smem;               // [240][258]  X^T: (n, c)
  __bf16* Tt   = (__bf16*)(smem + 123840);    // [240][146]  theta^T: (n, e)
  __bf16* PhT  = (__bf16*)(smem + 193920);    // [240][146]  phi^T: (m, e)
  __bf16* Aff  = (__bf16*)smem;               // [240][258]  (aliases Xn by phase)
  __bf16* Xs2  = (__bf16*)(smem + 123840);    // [256][242]  X: (c, m) (aliases Tt/PhT)
  __bf16* OutT = (__bf16*)smem;               // [240][258]  (row-aliases Aff, per-wave safe)

  const __bf16* thwB = wb + THW_OFF;          // [144][256]
  const __bf16* phwB = wb + PHW_OFF;          // [144][256]
  const __bf16* prwB = wb + PRW_OFF;          // [256][256]

  const int tid  = threadIdx.x;
  const int lane = tid & 31, wv = tid >> 5;
  const int l15  = lane & 15, hi = lane >> 4;

  const int blk = blockIdx.x;
  const int b   = blk / 196;
  const int wi  = blk % 196;
  const int nh  = wi / 14, nw = wi % 14;
  const float* xb = x + (size_t)b * C_DIM * HW * HW;

  // ---- P0: load reflect-padded window -> Xn[n][c] bf16; rows n>=225 zeroed ----
  for (int idx = tid; idx < 240 * C_DIM; idx += 256) {
    int n = idx % 240, c = idx / 240;
    float v = 0.f;
    if (n < N_VALID) {
      int i = n / WSZ, j = n % WSZ;
      int hp = nh * WSZ + i, wp = nw * WSZ + j;
      int h = (hp < HW) ? hp : (2 * HW - 2 - hp);
      int w = (wp < HW) ? wp : (2 * HW - 2 - wp);
      v = xb[(size_t)c * HW * HW + (size_t)h * HW + w];
    }
    Xn[(size_t)n * XN_S + c] = (__bf16)v;
  }
  __syncthreads();

  // ---- P1: Tt[n][e] = sum_c X[n][c]*thw[e][c] ; PhT[m][e] = sum_c phw[e][c]*X[m][c] ----
  for (int job = wv; job < NT * ETL * 2; job += 8) {
    v8f acc = zero8();
    if (job < NT * ETL) {
      int nT = job / ETL, eT = job % ETL;
      #pragma unroll
      for (int kk = 0; kk < KC; ++kk) {
        v16bf a  = load_a_lds_nm(Xn, XN_S, nT * 16, kk * 32);     // A = X^T (M=n,K=c)
        v16bf bb = load_b_gbfT(thwB, C_DIM, kk * 32, eT * 16);    // B = thw^T (K=c,N=e)
        acc = wmma_bf16(a, bb, acc);
      }
      store_d_bf16(Tt, TT_S, nT * 16, eT * 16, acc);
    } else {
      int j2 = job - NT * ETL;
      int eT = j2 / NT, mT = j2 % NT;
      #pragma unroll
      for (int kk = 0; kk < KC; ++kk) {
        v16bf a  = load_a_gbf(phwB, C_DIM, eT * 16, kk * 32);     // A = phw (M=e,K=c)
        v16bf bb = load_b_ldsT_nm(Xn, XN_S, kk * 32, mT * 16);    // B = X (K=c,N=m)
        acc = wmma_bf16(a, bb, acc);
      }
      store_d_bf16_T(PhT, TT_S, eT * 16, mT * 16, acc);           // store transposed: (m,e)
    }
  }
  __syncthreads();

  // ---- P2: scores + row softmax per 16-row tile; write Aff bf16 rows ----
  for (int rt = wv; rt < NT; rt += 8) {
    v8f acc[NT];
    #pragma unroll
    for (int mt = 0; mt < NT; ++mt) acc[mt] = zero8();
    #pragma unroll
    for (int kk = 0; kk < KE; ++kk) {
      v16bf a = load_a_lds_mk(Tt, TT_S, rt * 16, kk * 32, 144);   // A = theta^T (M=n,K=e)
      #pragma unroll
      for (int mt = 0; mt < NT; ++mt) {
        v16bf bb = load_b_ldsT_mk(PhT, TT_S, kk * 32, mt * 16, 144); // B = phi (K=e,N=m)
        acc[mt] = wmma_bf16(a, bb, acc[mt]);
      }
    }
    const float sc = 1.0f / 12.0f;  // 1/sqrt(144)
    #pragma unroll
    for (int mt = 0; mt < NT; ++mt) {
      bool valid = (mt * 16 + l15) < N_VALID;
      #pragma unroll
      for (int r = 0; r < 8; ++r)
        acc[mt][r] = valid ? acc[mt][r] * sc : -3.0e38f;
    }
    const int rowb = rt * 16 + 8 * hi;
    #pragma unroll
    for (int r = 0; r < 8; ++r) {
      float rm = -3.0e38f;
      #pragma unroll
      for (int mt = 0; mt < NT; ++mt) rm = fmaxf(rm, acc[mt][r]);
      #pragma unroll
      for (int s = 1; s < 16; s <<= 1) rm = fmaxf(rm, __shfl_xor(rm, s, 32));
      float rs = 0.f;
      #pragma unroll
      for (int mt = 0; mt < NT; ++mt) {
        float e = __expf(acc[mt][r] - rm);
        acc[mt][r] = e;
        rs += e;
      }
      #pragma unroll
      for (int s = 1; s < 16; s <<= 1) rs += __shfl_xor(rs, s, 32);
      float inv = 1.0f / rs;
      int row = rowb + r;
      #pragma unroll
      for (int mt = 0; mt < NT; ++mt)
        Aff[(size_t)row * XN_S + mt * 16 + l15] = (__bf16)(acc[mt][r] * inv);
      Aff[(size_t)row * XN_S + 240 + l15] = (__bf16)0.f;  // zero pad cols 240..255
    }
  }
  __syncthreads();

  // ---- reload X (bf16) -> Xs2[c][m] over dead Tt/PhT ----
  for (int idx = tid; idx < C_DIM * 240; idx += 256) {
    int m = idx % 240, c = idx / 240;
    float v = 0.f;
    if (m < N_VALID) {
      int i = m / WSZ, j = m % WSZ;
      int hp = nh * WSZ + i, wp = nw * WSZ + j;
      int h = (hp < HW) ? hp : (2 * HW - 2 - hp);
      int w = (wp < HW) ? wp : (2 * HW - 2 - wp);
      v = xb[(size_t)c * HW * HW + (size_t)h * HW + w];
    }
    Xs2[(size_t)c * X2_S + m] = (__bf16)v;
  }
  __syncthreads();

  // ---- P3 + P4 fused per row tile (per-wave row ownership => no cross-wave hazard) ----
  for (int rt = wv; rt < NT; rt += 8) {
    // P3: outT[n][c] = sum_m aff[n][m] * X[c][m]
    {
      v8f acc[CTL];
      #pragma unroll
      for (int ct = 0; ct < CTL; ++ct) acc[ct] = zero8();
      #pragma unroll
      for (int kk = 0; kk < KM; ++kk) {
        v16bf a = load_a_lds_nm(Aff, XN_S, rt * 16, kk * 32);          // A = aff (M=n,K=m)
        #pragma unroll
        for (int ct = 0; ct < CTL; ++ct) {
          v16bf bb = load_b_ldsT_mk(Xs2, X2_S, kk * 32, ct * 16, 240); // B = X^T (K=m,N=c)
          acc[ct] = wmma_bf16(a, bb, acc[ct]);
        }
      }
      #pragma unroll
      for (int ct = 0; ct < CTL; ++ct)
        store_d_bf16(OutT, XN_S, rt * 16, ct * 16, acc[ct]);           // overwrites own aff rows
    }
    // P4: y[o][n] = sum_c prw[o][c] * outT[n][c] ; residual add; coalesced store
    {
      v8f acc[CTL];
      #pragma unroll
      for (int ct = 0; ct < CTL; ++ct) acc[ct] = zero8();
      #pragma unroll
      for (int kk = 0; kk < KC; ++kk) {
        v16bf bb = load_b_ldsT_nm(OutT, XN_S, kk * 32, rt * 16);       // B = outT^T (K=c,N=n), shared
        #pragma unroll
        for (int ot = 0; ot < CTL; ++ot) {
          v16bf a = load_a_gbf(prwB, C_DIM, ot * 16, kk * 32);         // A = prw (M=o,K=c)
          acc[ot] = wmma_bf16(a, bb, acc[ot]);
        }
      }
      // lanes map to consecutive n (spatial w) -> coalesced global access
      int n = rt * 16 + l15;
      bool ok = false; int h = 0, w = 0;
      if (n < N_VALID) {
        int i = n / WSZ, j = n % WSZ;
        h = nh * WSZ + i; w = nw * WSZ + j;
        ok = (h < HW) && (w < HW);
      }
      const size_t hw_off = (size_t)h * HW + w;
      #pragma unroll
      for (int ot = 0; ot < CTL; ++ot) {
        #pragma unroll
        for (int r = 0; r < 8; ++r) {
          int o = ot * 16 + r + 8 * hi;
          if (ok) {
            size_t gi = ((size_t)(b * C_DIM + o)) * (HW * HW) + hw_off;
            out[gi] = xb[(size_t)o * (HW * HW) + hw_off] + acc[ot][r];
          }
        }
      }
    }
  }
}

extern "C" void kernel_launch(void* const* d_in, const int* in_sizes, int n_in,
                              void* d_out, int out_size, void* d_ws, size_t ws_size,
                              hipStream_t stream) {
  const float* x   = (const float*)d_in[0];
  const float* thw = (const float*)d_in[1];
  const float* phw = (const float*)d_in[2];
  const float* prw = (const float*)d_in[3];
  float* out = (float*)d_out;
  __bf16* wb = (__bf16*)d_ws;  // 278,528 bytes used

  hipLaunchKernelGGL(cvt_weights_kernel, dim3((W_TOTAL + 255) / 256), dim3(256), 0, stream,
                     thw, phw, prw, wb);
  hipLaunchKernelGGL(wnl_attn_kernel, dim3(4 * 14 * 14), dim3(256), 0, stream,
                     x, wb, out);
}
